// MoeModule_1159641170483
// MI455X (gfx1250) — matrix operand order repaired and checked by reading, not compiled, not run on hardware
//
#include <hip/hip_runtime.h>
#include <hip/hip_bf16.h>

typedef __bf16 v16bf __attribute__((ext_vector_type(16)));
typedef __bf16 v8bf  __attribute__((ext_vector_type(8)));
typedef float  v8f   __attribute__((ext_vector_type(8)));

#define NT   8192      // tokens = B*S
#define DD   1024      // model dim
#define NE   8         // experts
#define BM   64        // token tile
#define BN   64        // col tile
#define MT_PER_E (NT / BM)   // 128

// ---------------------------------------------------------------- zero out
__global__ void moe_zero(float* __restrict__ out, int n, int* __restrict__ counts) {
    int i = blockIdx.x * blockDim.x + threadIdx.x;
    if (i < n) out[i] = 0.0f;
    if (i < NE) counts[i] = 0;
}

// ------------------------------------------------------- x -> bf16 hi/lo
__global__ void moe_conv_x(const float* __restrict__ x,
                           __bf16* __restrict__ xh, __bf16* __restrict__ xl, int n) {
    int i = blockIdx.x * blockDim.x + threadIdx.x;
    if (i < n) {
        float f = x[i];
        __bf16 h = (__bf16)f;
        xh[i] = h;
        xl[i] = (__bf16)(f - (float)h);
    }
}

// ----------------- expert_W -> bf16 hi/lo, pre-swizzled to B-fragment layout
// Output element index: (((e*64 + nt)*32 + kb)*32 + lane)*16 + m
// holds W[e][kb*32 + (lane>>4)*16 + m][nt*16 + (lane&15)]
__global__ __launch_bounds__(256) void moe_conv_w(const float* __restrict__ W,
                                                  __bf16* __restrict__ wh,
                                                  __bf16* __restrict__ wl) {
    int bid = blockIdx.x;              // (e*64 + nt)*32 + kb
    int kb = bid & 31;
    int nt = (bid >> 5) & 63;
    int e  = bid >> 11;
    __shared__ float tile[32 * 16];    // [k_local][col]
    int tid = threadIdx.x;
    int kk  = tid >> 4;                // 0..15
    int col = tid & 15;
    size_t src = ((size_t)e * DD + (size_t)kb * 32) * DD + (size_t)nt * 16;
    tile[kk * 16 + col]        = W[src + (size_t)kk * DD + col];
    tile[(kk + 16) * 16 + col] = W[src + (size_t)(kk + 16) * DD + col];
    __syncthreads();
    // thread t writes output elements 2t, 2t+1 (lane = t>>3, m0 = (t&7)*2)
    int lane = tid >> 3;
    int m0   = (tid & 7) * 2;
    int kg   = lane >> 4;
    int c2   = lane & 15;
    float f0 = tile[(kg * 16 + m0) * 16 + c2];
    float f1 = tile[(kg * 16 + m0 + 1) * 16 + c2];
    __bf16 h0 = (__bf16)f0, h1 = (__bf16)f1;
    size_t o = (size_t)bid * 512 + (size_t)tid * 2;
    wh[o] = h0; wh[o + 1] = h1;
    wl[o]     = (__bf16)(f0 - (float)h0);
    wl[o + 1] = (__bf16)(f1 - (float)h1);
}

// ---------------------------------------------------------------- gating
__global__ __launch_bounds__(256) void moe_route(const float* __restrict__ x,
                                                 const float* __restrict__ gW,
                                                 const float* __restrict__ gb,
                                                 int* __restrict__ counts,
                                                 int* __restrict__ toklist,
                                                 float* __restrict__ wlist) {
    int t   = blockIdx.x;
    int tid = threadIdx.x;
    float acc[NE];
#pragma unroll
    for (int e = 0; e < NE; ++e) acc[e] = 0.0f;
    const float* xr = x + (size_t)t * DD;
    for (int d = tid; d < DD; d += 256) {
        float xv = xr[d];
        const float* gr = gW + (size_t)d * NE;
#pragma unroll
        for (int e = 0; e < NE; ++e) acc[e] += xv * gr[e];
    }
    // in-wave reduce (wave32)
#pragma unroll
    for (int e = 0; e < NE; ++e)
        for (int s = 16; s > 0; s >>= 1) acc[e] += __shfl_down(acc[e], s, 32);
    __shared__ float sred[8 * NE];
    int wid = tid >> 5, ln = tid & 31;
    if (ln == 0)
#pragma unroll
        for (int e = 0; e < NE; ++e) sred[wid * NE + e] = acc[e];
    __syncthreads();
    if (tid == 0) {
        float sc[NE], p[NE];
        float mx = -1e30f;
#pragma unroll
        for (int e = 0; e < NE; ++e) {
            float s = sred[e];
            for (int w = 1; w < 8; ++w) s += sred[w * NE + e];
            s = fmaxf(s + gb[e], 1e-4f);   // /TEMP(=1), clamp EPS
            sc[e] = s;
            mx = fmaxf(mx, s);
        }
        float sum = 0.0f;
#pragma unroll
        for (int e = 0; e < NE; ++e) { p[e] = __expf(sc[e] - mx); sum += p[e]; }
        float inv = 1.0f / sum;
#pragma unroll
        for (int e = 0; e < NE; ++e) p[e] *= inv;
        // top-2 (lowest index wins ties, matching lax.top_k)
        int i0 = 0;
        for (int e = 1; e < NE; ++e) if (p[e] > p[i0]) i0 = e;
        int i1 = -1;
        for (int e = 0; e < NE; ++e)
            if (e != i0 && (i1 < 0 || p[e] > p[i1])) i1 = e;
        float w0 = fmaxf(p[i0], 0.1f), w1 = fmaxf(p[i1], 0.1f);
        float tot = w0 + w1;
        w0 = w0 / tot * 0.5f;              // includes /K
        w1 = w1 / tot * 0.5f;
        int p0 = atomicAdd(&counts[i0], 1);
        toklist[i0 * NT + p0] = t; wlist[i0 * NT + p0] = w0;
        int p1 = atomicAdd(&counts[i1], 1);
        toklist[i1 * NT + p1] = t; wlist[i1 * NT + p1] = w1;
    }
}

// ------------------------------------------------- routed expert GEMM (3xBF16)
__global__ __launch_bounds__(256) void moe_gemm(const __bf16* __restrict__ xh,
                                                const __bf16* __restrict__ xl,
                                                const __bf16* __restrict__ wh,
                                                const __bf16* __restrict__ wl,
                                                const float* __restrict__ expb,
                                                const int* __restrict__ counts,
                                                const int* __restrict__ toklist,
                                                const float* __restrict__ wlist,
                                                float* __restrict__ out) {
    int e  = blockIdx.y / MT_PER_E;
    int mt = blockIdx.y % MT_PER_E;
    int cnt = counts[e];
    int row0 = mt * BM;
    if (row0 >= cnt) return;

    int tid = threadIdx.x;
    int lane = tid & 31;
    int w    = tid >> 5;      // 0..7
    int wm   = w & 3;         // token-row group (4)
    int wn   = w >> 2;        // col group (2)
    int kgroup = lane >> 4;
    int l15    = lane & 15;

    // token for A-fragment rows (A layout: lane&15 = row)
    int arow = row0 + wm * 16 + l15;
    int tokA = (arow < cnt) ? toklist[e * NT + arow] : 0;
    const __bf16* xhr = xh + (size_t)tokA * DD;
    const __bf16* xlr = xl + (size_t)tokA * DD;

    int nt0 = blockIdx.x * (BN / 16) + wn * 2;  // two N-fragments: nt0, nt0+1
    const __bf16* whb0 = wh + (size_t)(e * 64 + nt0) * 16384;
    const __bf16* wlb0 = wl + (size_t)(e * 64 + nt0) * 16384;
    const __bf16* whb1 = whb0 + 16384;
    const __bf16* wlb1 = wlb0 + 16384;

    v8f c0 = {}, c1 = {};
    for (int kb = 0; kb < DD / 32; ++kb) {
        int ka = kb * 32 + kgroup * 16;         // A: 16 contiguous k per lane
        v8bf a0h = *(const v8bf*)(xhr + ka);
        v8bf a1h = *(const v8bf*)(xhr + ka + 8);
        v8bf a0l = *(const v8bf*)(xlr + ka);
        v8bf a1l = *(const v8bf*)(xlr + ka + 8);
        v16bf Ah, Al;
#pragma unroll
        for (int i = 0; i < 8; ++i) {
            Ah[i] = a0h[i]; Ah[i + 8] = a1h[i];
            Al[i] = a0l[i]; Al[i + 8] = a1l[i];
        }
        size_t boff = (size_t)kb * 512 + (size_t)lane * 16;
        v16bf B0h = *(const v16bf*)(whb0 + boff);
        v16bf B0l = *(const v16bf*)(wlb0 + boff);
        v16bf B1h = *(const v16bf*)(whb1 + boff);
        v16bf B1l = *(const v16bf*)(wlb1 + boff);

        c0 = __builtin_amdgcn_wmma_f32_16x16x32_bf16(false, Ah, false, B0h, (short)0, c0, false, false);
        c1 = __builtin_amdgcn_wmma_f32_16x16x32_bf16(false, Ah, false, B1h, (short)0, c1, false, false);
        c0 = __builtin_amdgcn_wmma_f32_16x16x32_bf16(false, Ah, false, B0l, (short)0, c0, false, false);
        c1 = __builtin_amdgcn_wmma_f32_16x16x32_bf16(false, Ah, false, B1l, (short)0, c1, false, false);
        c0 = __builtin_amdgcn_wmma_f32_16x16x32_bf16(false, Al, false, B0h, (short)0, c0, false, false);
        c1 = __builtin_amdgcn_wmma_f32_16x16x32_bf16(false, Al, false, B1h, (short)0, c1, false, false);
    }

    // C layout: VGPR r -> row r + 8*(lane>>4), col = lane&15
    int colg0 = blockIdx.x * BN + wn * 32 + l15;
    int colg1 = colg0 + 16;
    float b0 = expb[e * DD + colg0];
    float b1 = expb[e * DD + colg1];
#pragma unroll
    for (int r = 0; r < 8; ++r) {
        int rowg = row0 + wm * 16 + kgroup * 8 + r;
        if (rowg < cnt) {
            int tok  = toklist[e * NT + rowg];
            float wt = wlist[e * NT + rowg];
            atomicAdd(&out[(size_t)tok * DD + colg0], wt * (c0[r] + b0));
            atomicAdd(&out[(size_t)tok * DD + colg1], wt * (c1[r] + b1));
        }
    }
}

// ---------------------------------------------------------------- launch
extern "C" void kernel_launch(void* const* d_in, const int* in_sizes, int n_in,
                              void* d_out, int out_size, void* d_ws, size_t ws_size,
                              hipStream_t stream) {
    const float* x    = (const float*)d_in[0];   // [4,2048,1024]
    const float* gW   = (const float*)d_in[1];   // [1024,8]
    const float* gb   = (const float*)d_in[2];   // [8]
    const float* eW   = (const float*)d_in[3];   // [8,1024,1024]
    const float* eb   = (const float*)d_in[4];   // [8,1024]
    float* out = (float*)d_out;                  // [4,2048,1024]

    char* ws = (char*)d_ws;
    size_t off = 0;
    __bf16* xh = (__bf16*)(ws + off); off += (size_t)NT * DD * 2;   // 16 MB
    __bf16* xl = (__bf16*)(ws + off); off += (size_t)NT * DD * 2;   // 16 MB
    __bf16* wh = (__bf16*)(ws + off); off += (size_t)NE * DD * DD * 2; // 16 MB
    __bf16* wl = (__bf16*)(ws + off); off += (size_t)NE * DD * DD * 2; // 16 MB
    int*   counts  = (int*)(ws + off);   off += 256;
    int*   toklist = (int*)(ws + off);   off += (size_t)NE * NT * 4;
    float* wlist   = (float*)(ws + off); off += (size_t)NE * NT * 4;

    int ntot = NT * DD;
    moe_zero<<<(ntot + 255) / 256, 256, 0, stream>>>(out, ntot, counts);
    moe_conv_x<<<(ntot + 255) / 256, 256, 0, stream>>>(x, xh, xl, ntot);
    moe_conv_w<<<NE * 64 * 32, 256, 0, stream>>>(eW, wh, wl);
    moe_route<<<NT, 256, 0, stream>>>(x, gW, gb, counts, toklist, wlist);
    moe_gemm<<<dim3(DD / BN, NE * MT_PER_E), 256, 0, stream>>>(
        xh, xl, wh, wl, eb, counts, toklist, wlist, out);
}
